// RuntimeAwareGraphAttention_30820685316574
// MI455X (gfx1250) — compile-verified, matrix operand order checked by reading, not run on hardware
//
#include <hip/hip_runtime.h>

// Reference: attention fwd returning (out, attn).
// B=4, H=16, L=S=2048, D=64, fp32 in/out.
// d_in: q, k, v, edge_mask_bool (1-byte bool), time_bias
// d_out: [out (B*H*L*D) | attn (B*H*L*S)] fp32, concatenated.

#define B_  4
#define H_  16
#define L_  2048
#define S_  2048
#define D_  64
#define BM  128   // L-rows per block (8 waves x 16 rows)
#define SC  32    // S-chunk (WMMA K for the PV matmul)

typedef __attribute__((ext_vector_type(16))) __bf16       v16bf;
typedef __attribute__((ext_vector_type(8)))  float        v8f;
typedef __attribute__((ext_vector_type(4)))  unsigned int u32x4;
typedef __attribute__((ext_vector_type(4)))  float        f32x4;

union V16U {
    u32x4          q[2];   // 32 bytes
    v16bf          v;
    unsigned short us[16];
};

__device__ __forceinline__ unsigned short f2bf(float f) {
    unsigned int u = __builtin_bit_cast(unsigned int, f);
    u += 0x7FFFu + ((u >> 16) & 1u);           // round-to-nearest-even
    return (unsigned short)(u >> 16);
}

__global__ __launch_bounds__(256)
void attn_fwd_kernel(const float* __restrict__ q,
                     const float* __restrict__ kp,
                     const float* __restrict__ vp,
                     const unsigned char* __restrict__ mask,
                     const float* __restrict__ bias,
                     float* __restrict__ out,
                     float* __restrict__ attn)
{
    // LDS staging (bf16). Row strides padded so every 16B A/B-operand read is
    // 16B-aligned and rows land on distinct bank groups.
    __shared__ __align__(16) unsigned short Kl[SC][72];      // [s][d]  4.5 KB
    __shared__ __align__(16) unsigned short Vl[D_][40];      // [d][s]  5.0 KB (transposed)
    __shared__ __align__(16) unsigned short Pl[8][16][40];   // per-wave P tile 10 KB

    const int b    = blockIdx.z;
    const int h    = blockIdx.y;
    const int bh   = b * H_ + h;
    const int lane = threadIdx.x & 31;
    const int w    = threadIdx.x >> 5;
    const int half = lane >> 4;          // half-wave select (WMMA layouts)
    const int l16  = lane & 15;
    const int baseK = half ? 8 : 0;      // lane's K-subgroup inside a 32-wide K
    const int row0 = blockIdx.x * BM + w * 16;

    // ---- Q A-operands (K=0..31 and K=32..63), pre-scaled by 1/sqrt(D) ----
    const float* qrow = q + (((size_t)bh * L_) + (row0 + l16)) * D_;
    V16U qa0, qa1;
#pragma unroll
    for (int i = 0; i < 8; ++i) {
        const int d0 = baseK + i;
        qa0.us[i]     = f2bf(qrow[d0]      * 0.125f);
        qa0.us[i + 8] = f2bf(qrow[d0 + 16] * 0.125f);
        qa1.us[i]     = f2bf(qrow[d0 + 32] * 0.125f);
        qa1.us[i + 8] = f2bf(qrow[d0 + 48] * 0.125f);
    }

    // Online softmax state: one row per (r, half): row = row0 + r + 8*half.
    float mrow[8], lrow[8];
#pragma unroll
    for (int r = 0; r < 8; ++r) { mrow[r] = -3.0e38f; lrow[r] = 0.0f; }

    // ---- cooperative K/V chunk loader: global fp32 -> LDS bf16 ----
    auto load_chunk = [&](int s0, bool loadV) {
#pragma unroll
        for (int p = 0; p < 2; ++p) {
            const int e  = (threadIdx.x + p * 256) * 4;   // 2048 elems per matrix
            const int sl = e >> 6;
            const int d0 = e & 63;
            const f32x4 kv = *(const f32x4*)(kp + (((size_t)bh * S_) + (s0 + sl)) * D_ + d0);
            Kl[sl][d0 + 0] = f2bf(kv.x);
            Kl[sl][d0 + 1] = f2bf(kv.y);
            Kl[sl][d0 + 2] = f2bf(kv.z);
            Kl[sl][d0 + 3] = f2bf(kv.w);
            if (loadV) {
                const f32x4 vv = *(const f32x4*)(vp + (((size_t)bh * S_) + (s0 + sl)) * D_ + d0);
                Vl[d0 + 0][sl] = f2bf(vv.x);    // transpose: [d][s]
                Vl[d0 + 1][sl] = f2bf(vv.y);
                Vl[d0 + 2][sl] = f2bf(vv.z);
                Vl[d0 + 3][sl] = f2bf(vv.w);
            }
        }
    };

    // ---- 16x16 score tile j (cols s0+16j .. +15): QK^T + bias + mask ----
    auto score_tile = [&](int s0, int j, float* sc) {
        const unsigned short* kr = &Kl[j * 16 + l16][0];
        V16U b0, b1;
        b0.q[0] = *(const u32x4*)(kr + baseK);
        b0.q[1] = *(const u32x4*)(kr + 16 + baseK);
        b1.q[0] = *(const u32x4*)(kr + 32 + baseK);
        b1.q[1] = *(const u32x4*)(kr + 48 + baseK);
        v8f c = {};
        c = __builtin_amdgcn_wmma_f32_16x16x32_bf16(false, qa0.v, false, b0.v,
                                                    (short)0, c, false, false);
        c = __builtin_amdgcn_wmma_f32_16x16x32_bf16(false, qa1.v, false, b1.v,
                                                    (short)0, c, false, false);
        const int colg = s0 + j * 16 + l16;
#pragma unroll
        for (int r = 0; r < 8; ++r) {
            const int rowg = row0 + r + 8 * half;
            const size_t off = (((size_t)b * L_) + rowg) * S_ + colg; // [B,1,L,S]
            float val = c[r] + bias[off];
            if (mask[off]) val = -1e9f;
            sc[r] = val;
        }
    };

    auto rmax16 = [&](float x) {
#pragma unroll
        for (int mk = 1; mk < 16; mk <<= 1) x = fmaxf(x, __shfl_xor(x, mk, 32));
        return x;
    };
    auto rsum16 = [&](float x) {
#pragma unroll
        for (int mk = 1; mk < 16; mk <<= 1) x += __shfl_xor(x, mk, 32);
        return x;
    };

    // ================= pass 1: row max / row sum =================
    for (int s0 = 0; s0 < S_; s0 += SC) {
        __syncthreads();
        load_chunk(s0, false);
        __syncthreads();
#pragma unroll
        for (int j = 0; j < 2; ++j) {
            float sc[8];
            score_tile(s0, j, sc);
#pragma unroll
            for (int r = 0; r < 8; ++r) {
                const float tm = rmax16(sc[r]);
                const float nm = fmaxf(mrow[r], tm);
                const float ts = rsum16(__expf(sc[r] - nm));
                lrow[r] = lrow[r] * __expf(mrow[r] - nm) + ts;
                mrow[r] = nm;
            }
        }
    }

    float rl[8];
#pragma unroll
    for (int r = 0; r < 8; ++r) rl[r] = 1.0f / lrow[r];

    // ================= pass 2: write attn, accumulate out =================
    v8f o[4] = {v8f{}, v8f{}, v8f{}, v8f{}};   // 16x64 output tile, f32

    for (int s0 = 0; s0 < S_; s0 += SC) {
        __syncthreads();
        load_chunk(s0, true);
        __syncthreads();
#pragma unroll
        for (int j = 0; j < 2; ++j) {
            float sc[8];
            score_tile(s0, j, sc);
#pragma unroll
            for (int r = 0; r < 8; ++r) {
                const int rowg = row0 + r + 8 * half;
                const float p = __expf(sc[r] - mrow[r]) * rl[r];
                attn[(((size_t)bh * L_) + rowg) * S_ + (s0 + j * 16 + l16)] = p;
                Pl[w][r + 8 * half][j * 16 + l16] = f2bf(p);   // stage for A-layout
            }
        }
        // P (16x32, A-operand) x V (32x64) -> accumulate 4 N-tiles
        V16U pa;
        const unsigned short* pr = &Pl[w][l16][0];
        pa.q[0] = *(const u32x4*)(pr + baseK);
        pa.q[1] = *(const u32x4*)(pr + 16 + baseK);
#pragma unroll
        for (int t = 0; t < 4; ++t) {
            V16U vb;
            const unsigned short* vr = &Vl[t * 16 + l16][0];
            vb.q[0] = *(const u32x4*)(vr + baseK);
            vb.q[1] = *(const u32x4*)(vr + 16 + baseK);
            o[t] = __builtin_amdgcn_wmma_f32_16x16x32_bf16(false, pa.v, false, vb.v,
                                                           (short)0, o[t], false, false);
        }
    }

    // ---- write out tile [16 x 64] ----
#pragma unroll
    for (int t = 0; t < 4; ++t) {
#pragma unroll
        for (int r = 0; r < 8; ++r) {
            const int rowg = row0 + r + 8 * half;
            out[(((size_t)bh * L_) + rowg) * D_ + t * 16 + l16] = o[t][r];
        }
    }
}

extern "C" void kernel_launch(void* const* d_in, const int* in_sizes, int n_in,
                              void* d_out, int out_size, void* d_ws, size_t ws_size,
                              hipStream_t stream) {
    (void)in_sizes; (void)n_in; (void)out_size; (void)d_ws; (void)ws_size;
    const float*         q    = (const float*)d_in[0];
    const float*         k    = (const float*)d_in[1];
    const float*         v    = (const float*)d_in[2];
    const unsigned char* mask = (const unsigned char*)d_in[3]; // jax bool = 1 byte
    const float*         bias = (const float*)d_in[4];

    float* out  = (float*)d_out;
    float* attn = out + (size_t)B_ * H_ * L_ * D_;   // tuple: (out, attn) concatenated

    dim3 grid(L_ / BM, H_, B_);
    dim3 block(256);
    hipLaunchKernelGGL(attn_fwd_kernel, grid, block, 0, stream,
                       q, k, v, mask, bias, out, attn);
}
